// MoNet_40699110097229
// MI455X (gfx1250) — compile-verified, matrix operand order checked by reading, not run on hardware
//
#include <hip/hip_runtime.h>
#include <hip/hip_bf16.h>

typedef __attribute__((ext_vector_type(2)))  float     v2f;
typedef __attribute__((ext_vector_type(8)))  float     v8f;
typedef __attribute__((ext_vector_type(16))) _Float16  v16h;

#define MN 50000
#define ME 800000
#define MIN_DIM 128
#define MHID 64
#define MK 3
#define MNCLS 16
#define MNL 3

// ---------------- generic helpers ----------------

__global__ void zero4_kernel(float4* __restrict__ p, long long n4) {
    long long i = (long long)blockIdx.x * blockDim.x + threadIdx.x;
    if (i < n4) p[i] = make_float4(0.f, 0.f, 0.f, 0.f);
}

// narrow int64 edge index -> packed int2 (r,c); read once, reused by all passes
__global__ void idx_kernel(const long long* __restrict__ row,
                           const long long* __restrict__ col,
                           int2* __restrict__ idx, int E) {
    int e = blockIdx.x * blockDim.x + threadIdx.x;
    if (e >= E) return;
    idx[e] = make_int2((int)row[e], (int)col[e]);
}

__global__ void deg_kernel(const int2* __restrict__ idx,
                           float* __restrict__ deg_r, float* __restrict__ deg_c, int E) {
    int e = blockIdx.x * blockDim.x + threadIdx.x;
    if (e >= E) return;
    int2 rc = idx[e];
    atomicAdd(deg_r + rc.x, 1.0f);
    atomicAdd(deg_c + rc.y, 1.0f);
}

__global__ void pseudo_kernel(const int2* __restrict__ idx,
                              const float* __restrict__ deg_r,
                              const float* __restrict__ deg_c,
                              float2* __restrict__ pseudo, int E) {
    int e = blockIdx.x * blockDim.x + threadIdx.x;
    if (e >= E) return;
    int2 rc = idx[e];
    pseudo[e] = make_float2(rsqrtf(deg_r[rc.x] + 1.0f), rsqrtf(deg_c[rc.y] + 1.0f));
}

// per-edge, per-layer gaussian kernel weights: gauss[e][k]
__global__ void gauss_kernel(const float2* __restrict__ pseudo,
                             const float* __restrict__ ppw,   // [2][2]
                             const float* __restrict__ ppb,   // [2]
                             const float* __restrict__ mu,    // [3][2]
                             const float* __restrict__ isg,   // [3][2]
                             float* __restrict__ gauss, int E) {
    int e = blockIdx.x * blockDim.x + threadIdx.x;
    if (e >= E) return;
    float2 ps = pseudo[e];
    float p0 = tanhf(ps.x * ppw[0] + ps.y * ppw[2] + ppb[0]);
    float p1 = tanhf(ps.x * ppw[1] + ps.y * ppw[3] + ppb[1]);
#pragma unroll
    for (int k = 0; k < MK; ++k) {
        float d0 = (p0 - mu[2 * k + 0]) * isg[2 * k + 0];
        float d1 = (p1 - mu[2 * k + 1]) * isg[2 * k + 1];
        gauss[3 * (long long)e + k] = expf(-0.5f * (d0 * d0 + d1 * d1));
    }
}

// ---------------- WMMA GEMM: Out[M,Nd] = A[M,Kd] @ W[Kd,Nd] + bias ----------------
// One wave computes one 16x16 output tile. M, Nd multiples of 16; Kd multiple of 4.
// Wave-uniform early exit keeps EXEC all-1s for WMMA (ISA requirement).

__global__ __launch_bounds__(256)
void gemm_wmma(const float* __restrict__ A, const float* __restrict__ W,
               const float* __restrict__ bias, float* __restrict__ Out,
               int M, int Kd, int Nd) {
    int wave = blockIdx.x * (blockDim.x >> 5) + (threadIdx.x >> 5);
    int lane = threadIdx.x & 31;
    int ntN = Nd >> 4;
    int ntiles = (M >> 4) * ntN;
    if (wave >= ntiles) return;           // uniform over the wave
    int tm = (wave / ntN) << 4;           // output row tile base
    int tn = (wave % ntN) << 4;           // output col tile base
    int half = lane >> 4;
    int mn = lane & 15;

    v8f acc = {0.f, 0.f, 0.f, 0.f, 0.f, 0.f, 0.f, 0.f};

#if __has_builtin(__builtin_amdgcn_wmma_f32_16x16x4_f32)
    // f32 A 16x4 layout: lanes 0-15 hold M rows, VGPR0={K=0|K=2}, VGPR1={K=1|K=3}
    const float* arow = A + (size_t)(tm + mn) * Kd + 2 * half;
    const float* bcol = W + (size_t)(2 * half) * Nd + tn + mn;
#pragma unroll 4
    for (int k0 = 0; k0 < Kd; k0 += 4) {
        v2f a, b;
        a.x = arow[k0];
        a.y = arow[k0 + 1];
        b.x = bcol[(size_t)k0 * Nd];
        b.y = bcol[(size_t)(k0 + 1) * Nd];
        acc = __builtin_amdgcn_wmma_f32_16x16x4_f32(
            false, a, false, b, (short)0, acc, false, false);
    }
#else
    // fallback: convert to f16 and use the codegen-confirmed 16x16x32 f16 WMMA
    const float* arow = A + (size_t)(tm + mn) * Kd;
    for (int k0 = 0; k0 < Kd; k0 += 32) {
        v16h a, b;
#pragma unroll
        for (int j = 0; j < 16; ++j) {
            int v = j >> 1, p = j & 1;
            int kk = ((v & 4) << 2) + (half << 3) + ((v & 3) << 1) + p;
            a[j] = (_Float16)arow[k0 + kk];
            b[j] = (_Float16)W[(size_t)(k0 + kk) * Nd + tn + mn];
        }
        acc = __builtin_amdgcn_wmma_f32_16x16x32_f16(
            false, a, false, b, (short)0, acc, false, false);
    }
#endif

    float bv = bias[tn + mn];
#pragma unroll
    for (int v = 0; v < 8; ++v) {
        // D layout: VGPR v = row (v | v+8), lane = col
        Out[(size_t)(tm + v + 8 * half) * Nd + tn + mn] = acc[v] + bv;
    }
}

// ---------------- edge gather / kernel-weight / scatter-add ----------------
// thread = (edge, 4-channel group); hp is [N, K*Cout]; three b128 gathers per thread,
// f32 atomic scatter (everything L2-resident on the 192MB L2).

__global__ void edge_kernel(const int2* __restrict__ idx,
                            const float* __restrict__ gauss,
                            const float* __restrict__ hp,
                            float* __restrict__ agg,
                            int E, int Cout, int logG) {
    long long gid = (long long)blockIdx.x * blockDim.x + threadIdx.x;
    int groups = Cout >> 2;               // 4-channel groups per edge
    if (gid >= ((long long)E << logG)) return;
    int e  = (int)(gid >> logG);
    int c4 = ((int)gid & (groups - 1)) << 2;
    int2 rc = idx[e];
    const float* g    = gauss + 3 * (size_t)e;
    const float* base = hp + (size_t)rc.x * (3 * Cout) + c4;
    float4 h0 = *(const float4*)(base);
    float4 h1 = *(const float4*)(base + Cout);
    float4 h2 = *(const float4*)(base + 2 * Cout);
    float g0 = g[0], g1 = g[1], g2 = g[2];
    float4 m;
    m.x = g0 * h0.x + g1 * h1.x + g2 * h2.x;
    m.y = g0 * h0.y + g1 * h1.y + g2 * h2.y;
    m.z = g0 * h0.z + g1 * h1.z + g2 * h2.z;
    m.w = g0 * h0.w + g1 * h1.w + g2 * h2.w;
    float* out = agg + (size_t)rc.y * Cout + c4;
    atomicAdd(out + 0, m.x);
    atomicAdd(out + 1, m.y);
    atomicAdd(out + 2, m.z);
    atomicAdd(out + 3, m.w);
}

// ---------------- batch-norm (train stats) ----------------

__global__ __launch_bounds__(256)
void bn_reduce(const float* __restrict__ agg, float* __restrict__ sums, int N, int C) {
    int c = threadIdx.x % C;
    int sub = threadIdx.x / C;
    int nsub = blockDim.x / C;
    float s = 0.f, s2 = 0.f;
    for (int n = blockIdx.x * nsub + sub; n < N; n += gridDim.x * nsub) {
        float v = agg[(size_t)n * C + c];
        s += v; s2 += v * v;
    }
    __shared__ float sh[512];
    sh[threadIdx.x] = s;
    sh[256 + threadIdx.x] = s2;
    __syncthreads();
    if (sub == 0) {
        for (int i = 1; i < nsub; ++i) {
            s  += sh[i * C + c];
            s2 += sh[256 + i * C + c];
        }
        atomicAdd(&sums[c], s);
        atomicAdd(&sums[C + c], s2);
    }
}

__global__ void bn_apply(const float* __restrict__ agg, const float* __restrict__ sums,
                         const float* __restrict__ g, const float* __restrict__ b,
                         const float* __restrict__ hres, float* __restrict__ out,
                         int N, int C, int residual) {
    long long t4 = (long long)blockIdx.x * blockDim.x + threadIdx.x;
    long long total4 = (long long)N * C / 4;
    if (t4 >= total4) return;
    long long t = t4 * 4;
    int c = (int)(t % C);
    float invN = 1.0f / (float)N;
    float4 a  = *(const float4*)(agg + t);
    float4 r;
    float vv[4] = {a.x, a.y, a.z, a.w};
    float ro[4];
#pragma unroll
    for (int j = 0; j < 4; ++j) {
        int cj = c + j;
        float mean = sums[cj] * invN;
        float var  = sums[C + cj] * invN - mean * mean;
        float v = (vv[j] - mean) * rsqrtf(var + 1e-5f) * g[cj] + b[cj];
        ro[j] = fmaxf(v, 0.0f);
    }
    r = make_float4(ro[0], ro[1], ro[2], ro[3]);
    if (residual) {
        float4 hh = *(const float4*)(hres + t);
        r.x += hh.x; r.y += hh.y; r.z += hh.z; r.w += hh.w;
    }
    *(float4*)(out + t) = r;
}

// ---------------- orchestration ----------------

extern "C" void kernel_launch(void* const* d_in, const int* in_sizes, int n_in,
                              void* d_out, int out_size, void* d_ws, size_t ws_size,
                              hipStream_t stream) {
    const int N = MN, E = ME, IN_DIM = MIN_DIM, HID = MHID, NCLS = MNCLS;

    const float* feature = (const float*)d_in[0];
    const float* emb_w   = (const float*)d_in[1];
    const float* emb_b   = (const float*)d_in[2];
    const float* fc_w    = (const float*)d_in[3];   // [3,64,192]
    const float* fc_b    = (const float*)d_in[4];   // [3,192]
    const float* mu      = (const float*)d_in[5];   // [3,3,2]
    const float* isg     = (const float*)d_in[6];   // [3,3,2]
    const float* pp_w    = (const float*)d_in[7];   // [3,2,2]
    const float* pp_b    = (const float*)d_in[8];   // [3,2]
    const float* bn_g    = (const float*)d_in[9];   // [3,64]
    const float* bn_b    = (const float*)d_in[10];  // [3,64]
    const float* fc_w_l  = (const float*)d_in[11];  // [64,48]
    const float* fc_b_l  = (const float*)d_in[12];  // [48]
    const float* mu_l    = (const float*)d_in[13];  // [3,2]
    const float* isg_l   = (const float*)d_in[14];  // [3,2]
    const float* pp_w_l  = (const float*)d_in[15];  // [2,2]
    const float* pp_b_l  = (const float*)d_in[16];  // [2]
    const float* bn_g_l  = (const float*)d_in[17];  // [16]
    const float* bn_b_l  = (const float*)d_in[18];  // [16]
    const long long* eidx = (const long long*)d_in[19];
    const long long* row = eidx;
    const long long* col = eidx + E;
    (void)in_sizes; (void)n_in; (void)out_size; (void)ws_size;

    // workspace carving (all L2-resident: ~87 MB total vs 192 MB L2)
    char* ws = (char*)d_ws;
    size_t off = 0;
    auto carve = [&](size_t bytes) -> void* {
        void* p = ws + off;
        off += (bytes + 255) & ~(size_t)255;
        return p;
    };
    float*  deg    = (float*)carve((size_t)2 * N * 4);          // deg_r | deg_c
    float*  deg_r  = deg;
    float*  deg_c  = deg + N;
    int2*   idx    = (int2*)carve((size_t)E * 8);               // packed (r,c) int32
    float2* pseudo = (float2*)carve((size_t)E * 8);             // [E,2]
    float*  gauss  = (float*)carve((size_t)3 * E * 4);          // [E,3]
    float*  h      = (float*)carve((size_t)N * HID * 4);        // [N,64]
    float*  hp     = (float*)carve((size_t)N * 3 * HID * 4);    // [N,192]
    float*  agg    = (float*)carve((size_t)N * HID * 4);        // [N,64]
    float*  sums   = (float*)carve((size_t)2 * HID * 4);        // mean/var accum

    const int TB = 256;
    auto cdiv = [](long long a, long long b) { return (int)((a + b - 1) / b); };

    // indices, degrees, pseudo coordinates
    idx_kernel<<<cdiv(E, TB), TB, 0, stream>>>(row, col, idx, E);
    zero4_kernel<<<cdiv(2LL * N / 4, TB), TB, 0, stream>>>((float4*)deg, 2LL * N / 4);
    deg_kernel<<<cdiv(E, TB), TB, 0, stream>>>(idx, deg_r, deg_c, E);
    pseudo_kernel<<<cdiv(E, TB), TB, 0, stream>>>(idx, deg_r, deg_c, pseudo, E);

    // embedding: h = feature @ emb_w + emb_b   [50000,128]x[128,64]
    {
        int tiles = (N / 16) * (HID / 16);
        gemm_wmma<<<cdiv(tiles, 8), 256, 0, stream>>>(feature, emb_w, emb_b, h,
                                                      N, IN_DIM, HID);
    }

    // hidden GMM layers (residual)
    for (int i = 0; i < MNL; ++i) {
        gauss_kernel<<<cdiv(E, TB), TB, 0, stream>>>(pseudo,
                                                     pp_w + i * 4, pp_b + i * 2,
                                                     mu + i * 6, isg + i * 6,
                                                     gauss, E);
        int tiles = (N / 16) * ((3 * HID) / 16);
        gemm_wmma<<<cdiv(tiles, 8), 256, 0, stream>>>(h, fc_w + (size_t)i * HID * 3 * HID,
                                                      fc_b + i * 3 * HID, hp,
                                                      N, HID, 3 * HID);
        zero4_kernel<<<cdiv((long long)N * HID / 4, TB), TB, 0, stream>>>(
            (float4*)agg, (long long)N * HID / 4);
        zero4_kernel<<<1, TB, 0, stream>>>((float4*)sums, 2 * HID / 4);
        edge_kernel<<<cdiv((long long)E * (HID / 4), TB), TB, 0, stream>>>(
            idx, gauss, hp, agg, E, HID, 4);
        bn_reduce<<<256, 256, 0, stream>>>(agg, sums, N, HID);
        bn_apply<<<cdiv((long long)N * HID / 4, TB), TB, 0, stream>>>(
            agg, sums, bn_g + i * HID, bn_b + i * HID, h, h, N, HID, 1);
    }

    // classifier layer (no residual) -> d_out [50000,16]
    {
        gauss_kernel<<<cdiv(E, TB), TB, 0, stream>>>(pseudo, pp_w_l, pp_b_l,
                                                     mu_l, isg_l, gauss, E);
        int tiles = (N / 16) * ((3 * NCLS) / 16);
        gemm_wmma<<<cdiv(tiles, 8), 256, 0, stream>>>(h, fc_w_l, fc_b_l, hp,
                                                      N, HID, 3 * NCLS);
        zero4_kernel<<<cdiv((long long)N * NCLS / 4, TB), TB, 0, stream>>>(
            (float4*)agg, (long long)N * NCLS / 4);
        zero4_kernel<<<1, TB, 0, stream>>>((float4*)sums, 2 * NCLS / 4);
        edge_kernel<<<cdiv((long long)E * (NCLS / 4), TB), TB, 0, stream>>>(
            idx, gauss, hp, agg, E, NCLS, 2);
        bn_reduce<<<256, 256, 0, stream>>>(agg, sums, N, NCLS);
        bn_apply<<<cdiv((long long)N * NCLS / 4, TB), TB, 0, stream>>>(
            agg, sums, bn_g_l, bn_b_l, h, (float*)d_out, N, NCLS, 0);
    }
}
